// Attention_21165598835147
// MI455X (gfx1250) — matrix operand order repaired and checked by reading, not compiled, not run on hardware
//
#include <hip/hip_runtime.h>
#include <hip/hip_bf16.h>

// ---- problem constants (reference: T,N,C=(256,64,256), mV=64, mT=32) ----
#define T_   256
#define N_   64
#define C_   256
#define S_   2048      // mV*mT
#define MT_  32
#define MV_  64
#define NEGV (-1e15f)

#define HAS_ASYNC_LDS __has_builtin(__builtin_amdgcn_global_load_async_to_lds_b128)
#define HAS_ASYNCCNT  __has_builtin(__builtin_amdgcn_s_wait_asynccnt)

typedef __attribute__((ext_vector_type(16))) __bf16       v16bf;
typedef __attribute__((ext_vector_type(8)))  float        v8f;
typedef __attribute__((ext_vector_type(4)))  unsigned int u32x4;
typedef __attribute__((ext_vector_type(4)))  int          v4i;
typedef __attribute__((address_space(1))) v4i*            v4i_gptr;   // global
typedef __attribute__((address_space(3))) v4i*            v4i_lptr;   // LDS

union FragB {
    v16bf  v;
    u32x4  q[2];
    __bf16 h[16];
};

__device__ __forceinline__ v8f wmma_bf16(v16bf a, v16bf b, v8f c) {
    return __builtin_amdgcn_wmma_f32_16x16x32_bf16(false, a, false, b, (short)0, c,
                                                   false, false);
}

// 16x16 score tile: A = Q rows (bf16 row-major), B = K rows (bf16 row-major, used as K^T)
// qrow/krow are byte pointers to this lane's row (row index = lane%16), g = lane/16.
__device__ __forceinline__ v8f score_tile(const char* qrow, const char* krow, int g) {
    v8f acc = {};
#pragma unroll
    for (int cc = 0; cc < 8; ++cc) {           // K = 256 in chunks of 32
        FragB a, b;
        a.q[0] = *(const u32x4*)(qrow + cc * 64 + g * 32);
        a.q[1] = *(const u32x4*)(qrow + cc * 64 + g * 32 + 16);
        b.q[0] = *(const u32x4*)(krow + cc * 64 + g * 32);
        b.q[1] = *(const u32x4*)(krow + cc * 64 + g * 32 + 16);
        acc = wmma_bf16(a.v, b.v, acc);
    }
    return acc;
}

// fp32 source -> bf16 A fragment (lane row, chunk layout of 16x32 bf16 A matrix)
__device__ __forceinline__ void load_a_f32(FragB& a, const float* row, int cc, int g) {
    const float* p0 = row + cc * 32 + g * 16;
    const float* p1 = row + cc * 32 + g * 16 + 8;
#pragma unroll
    for (int i = 0; i < 8; ++i) {
        a.h[i]     = (__bf16)p0[i];
        a.h[8 + i] = (__bf16)p1[i];
    }
}

// fp32 weight row -> bf16 B fragment (lane holds output column = weight row, contiguous K)
__device__ __forceinline__ void load_b_f32(FragB& b, const float* row, int cc, int g) {
    const float* p = row + cc * 32 + g * 16;
#pragma unroll
    for (int i = 0; i < 16; ++i) b.h[i] = (__bf16)p[i];
}

// ---------------- Kernel 1: Q = Pn @ Wq^T + bq  ->  bf16 Qb[n*T+t][c] ----------------
__global__ __launch_bounds__(128) void proj_q_kernel(const float* __restrict__ P,
                                                     const float* __restrict__ Wq,
                                                     const float* __restrict__ bq,
                                                     __hip_bfloat16* __restrict__ Qb) {
    const int wv = threadIdx.x >> 5, lane = threadIdx.x & 31;
    const int j = lane & 15, g = lane >> 4;
    const int tile    = blockIdx.x * 4 + wv;    // 1024 row-tiles * 16 col-tiles
    const int rowTile = tile >> 4;
    const int colTile = tile & 15;

    const int arow = rowTile * 16 + j;          // flat n*T + t
    const int n = arow >> 8, t = arow & 255;
    const float* src  = P  + ((size_t)t * N_ + n) * C_;          // Pn[n][t][:] = P[t][n][:]
    const float* wrow = Wq + (size_t)(colTile * 16 + j) * C_;

    v8f acc = {};
#pragma unroll
    for (int cc = 0; cc < 8; ++cc) {
        FragB a, b;
        load_a_f32(a, src, cc, g);
        load_b_f32(b, wrow, cc, g);
        acc = wmma_bf16(a.v, b.v, acc);
    }
    const float bias = bq[colTile * 16 + j];
#pragma unroll
    for (int r = 0; r < 8; ++r) {
        const int drow = rowTile * 16 + r + 8 * g;
        Qb[(size_t)drow * C_ + colTile * 16 + j] = __float2bfloat16(acc[r] + bias);
    }
}

// ------------- Kernel 2: K = Mn @ Wk^T + bk (row-major), V -> transposed Vt -------------
__global__ __launch_bounds__(128) void proj_kv_kernel(const float* __restrict__ Mm,
                                                      const float* __restrict__ Wk,
                                                      const float* __restrict__ bk,
                                                      const float* __restrict__ Wv,
                                                      const float* __restrict__ bv,
                                                      __hip_bfloat16* __restrict__ Kb,
                                                      __hip_bfloat16* __restrict__ Vtb) {
    const int wv = threadIdx.x >> 5, lane = threadIdx.x & 31;
    const int j = lane & 15, g = lane >> 4;
    const int tile    = blockIdx.x * 4 + wv;    // 8192 row-tiles * 16 col-tiles
    const int rowTile = tile >> 4;
    const int colTile = tile & 15;

    const int arow = rowTile * 16 + j;          // flat n*S + s
    const int n = arow >> 11, s = arow & 2047;
    const int v = s >> 5, tt = s & 31;
    // Mn[n][v*32+tt][:] = M[tt][n*64+v][:]
    const float* src  = Mm + ((size_t)tt * (N_ * MV_) + (size_t)n * MV_ + v) * C_;
    const float* wkr  = Wk + (size_t)(colTile * 16 + j) * C_;
    const float* wvr  = Wv + (size_t)(colTile * 16 + j) * C_;

    v8f accK = {}, accV = {};
#pragma unroll
    for (int cc = 0; cc < 8; ++cc) {
        FragB a, b;
        load_a_f32(a, src, cc, g);
        load_b_f32(b, wkr, cc, g);
        accK = wmma_bf16(a.v, b.v, accK);
        load_b_f32(b, wvr, cc, g);
        accV = wmma_bf16(a.v, b.v, accV);
    }
    const int   c  = colTile * 16 + j;
    const float bK = bk[c], bV = bv[c];

#pragma unroll
    for (int r = 0; r < 8; ++r) {               // K row-major store
        const int drow = rowTile * 16 + r + 8 * g;
        Kb[(size_t)drow * C_ + c] = __float2bfloat16(accK[r] + bK);
    }
    {   // V transposed store: per lane 8 consecutive s -> one b128 store
        FragB tmp;
#pragma unroll
        for (int r = 0; r < 8; ++r) tmp.h[r] = (__bf16)(accV[r] + bV);
        const int nn    = (rowTile * 16) >> 11;
        const int sbase = (rowTile * 16) & 2047;
        *(u32x4*)(Vtb + ((size_t)nn * C_ + c) * S_ + sbase + 8 * g) = tmp.q[0];
    }
}

// --------------- Kernel 3: two-pass online-softmax attention, WMMA everywhere ---------------
// Block = 4 waves, each owning a 16-row t-strip of the same head n.  The V^T tile for the
// current v-group (256c x 32s bf16 = 16KB) is shared by all 4 waves, so it is staged into
// LDS once per block with async global->LDS copies (ASYNCcnt), double-buffered so the next
// tile streams in while the current one feeds the out-WMMAs.
__global__ __launch_bounds__(128) void attn_kernel(const __hip_bfloat16* __restrict__ Qb,
                                                   const __hip_bfloat16* __restrict__ Kb,
                                                   const __hip_bfloat16* __restrict__ Vtb,
                                                   const unsigned char* __restrict__ mask,
                                                   float* __restrict__ out,
                                                   float* __restrict__ att_part) {
    __shared__ __align__(16) unsigned short abuf_all[4][16][32];  // per-wave attn tile (bf16)
    __shared__ __align__(16) unsigned short vbuf[2][C_][32];      // double-buffered V^T tile

    const int tid = threadIdx.x;
    const int wv = tid >> 5, lane = tid & 31;
    const int j = lane & 15, g = lane >> 4;
    const int n  = blockIdx.x;
    const int t0 = blockIdx.y * 64 + wv * 16;

    const char* Qrow  = (const char*)(Qb + ((size_t)n * T_ + t0 + j) * C_);
    const char* Kbase = (const char*)(Kb + (size_t)n * S_ * C_);
    const char* Vbase = (const char*)(Vtb + (size_t)n * C_ * S_);
    const unsigned char* mrow = mask + n * MV_;

    // Per-lane online softmax state: 8 t-rows x 2 tile-parities (tt = j and j+16).
    float m0[8], d0[8], m1[8], d1[8];
#pragma unroll
    for (int r = 0; r < 8; ++r) { m0[r] = NEGV; d0[r] = 0.f; m1[r] = NEGV; d1[r] = 0.f; }

    // ---------------- pass 1: running max / denominator ----------------
    for (int v = 0; v < MV_; ++v) {
        if (mrow[v]) {
            const char* k0 = Kbase + ((size_t)(32 * v + j)) * (C_ * 2);
            const char* k1 = Kbase + ((size_t)(32 * v + 16 + j)) * (C_ * 2);
            v8f s0 = score_tile(Qrow, k0, g);
            v8f s1 = score_tile(Qrow, k1, g);
#pragma unroll
            for (int r = 0; r < 8; ++r) {
                float x = s0[r], mo = m0[r], mn = fmaxf(mo, x);
                d0[r] = d0[r] * __expf(mo - mn) + __expf(x - mn); m0[r] = mn;
                x = s1[r]; mo = m1[r]; mn = fmaxf(mo, x);
                d1[r] = d1[r] * __expf(mo - mn) + __expf(x - mn); m1[r] = mn;
            }
        } else {  // whole tile masked: score = NEG for every element of this v-group
#pragma unroll
            for (int r = 0; r < 8; ++r) {
                d0[r] += __expf(NEGV - m0[r]);
                d1[r] += __expf(NEGV - m1[r]);
            }
        }
    }
    float rd0[8], rd1[8];
#pragma unroll
    for (int r = 0; r < 8; ++r) {
        rd0[r] = __builtin_amdgcn_rcpf(d0[r]);
        rd1[r] = __builtin_amdgcn_rcpf(d1[r]);
    }

    // ---------------- pass 2: attn -> LDS transpose -> attn @ V ----------------
    v8f oacc[16];
    v8f zero = {};
#pragma unroll
    for (int ct = 0; ct < 16; ++ct) oacc[ct] = zero;

    unsigned short (*abuf)[32] = abuf_all[wv];

    // Stage the V^T tile for group v into vbuf[buf]: 1024 x 16B chunks, 8 per thread.
    auto stage_v = [&](int v, int buf) {
#pragma unroll
        for (int i = 0; i < 8; ++i) {
            const int idx = i * 128 + tid;                 // 0..1023
            const int c = idx >> 2, q = idx & 3;           // row c, 16B chunk q
            const char* gp = Vbase + (size_t)c * (S_ * 2) + v * 64 + q * 16;
            char* lp = (char*)&vbuf[buf][0][0] + c * 64 + q * 16;
#if HAS_ASYNC_LDS
            __builtin_amdgcn_global_load_async_to_lds_b128(
                (v4i_gptr)(void*)gp, (v4i_lptr)(void*)lp, 0, 0);
#else
            *(u32x4*)lp = *(const u32x4*)gp;
#endif
        }
    };

    stage_v(0, 0);

    for (int v = 0; v < MV_; ++v) {
        const bool last = (v + 1 >= MV_);
        if (!last) stage_v(v + 1, (v + 1) & 1);     // prefetch next tile into other buffer

        const bool ok = mrow[v] != 0;
        v8f s0 = {}, s1 = {};
        if (ok) {
            const char* k0 = Kbase + ((size_t)(32 * v + j)) * (C_ * 2);
            const char* k1 = Kbase + ((size_t)(32 * v + 16 + j)) * (C_ * 2);
            s0 = score_tile(Qrow, k0, g);
            s1 = score_tile(Qrow, k1, g);
        }
        float p = 0.f;
#pragma unroll
        for (int r = 0; r < 8; ++r) {
            const float x0 = ok ? s0[r] : NEGV;
            const float x1 = ok ? s1[r] : NEGV;
            const float a0 = __expf(x0 - m0[r]) * rd0[r];
            const float a1 = __expf(x1 - m1[r]) * rd1[r];
            p += a0 + a1;
            const int row = r + 8 * g;          // D-layout row for this lane
            const __bf16 b0 = (__bf16)a0, b1 = (__bf16)a1;
            abuf[row][j]      = __builtin_bit_cast(unsigned short, b0);
            abuf[row][16 + j] = __builtin_bit_cast(unsigned short, b1);
        }
        asm volatile("s_wait_dscnt 0x0" ::: "memory");  // D-tile stores visible before A-frag loads

        FragB af;                                        // attn 16x32 A-fragment from LDS
        const char* ap = (const char*)abuf + j * 64;
        af.q[0] = *(const u32x4*)(ap + g * 32);
        af.q[1] = *(const u32x4*)(ap + g * 32 + 16);

        // wait for this wave's copies of tile v (next tile's 8 copies may stay in flight),
        // then barrier so every wave's slice of the tile is visible.
#if HAS_ASYNC_LDS && HAS_ASYNCCNT
        if (!last) __builtin_amdgcn_s_wait_asynccnt(8);
        else       __builtin_amdgcn_s_wait_asynccnt(0);
#endif
        __syncthreads();

        const char* vtile = (const char*)&vbuf[v & 1][0][0];
#pragma unroll
        for (int ct = 0; ct < 16; ++ct) {               // out[:, ct*16 .. +15] += attn @ V
            FragB bfv;
            const char* vr = vtile + (ct * 16 + j) * 64 + g * 32;
            bfv.q[0] = *(const u32x4*)vr;
            bfv.q[1] = *(const u32x4*)(vr + 16);
            oacc[ct] = wmma_bf16(af.v, bfv.v, oacc[ct]);
        }

        // att partial: mean over this strip's 16 t and all 32 tt for group v
        p *= (1.0f / 8192.0f);                           // 1/(T*mT)
#pragma unroll
        for (int off = 16; off > 0; off >>= 1) p += __shfl_xor(p, off, 32);
        if (lane == 0) att_part[(size_t)(t0 >> 4) * (N_ * MV_) + n * MV_ + v] = p;

        __syncthreads();   // all waves done reading vbuf[v&1] before it is re-staged
    }

    // ---------------- store out (T, N, C) ----------------
#pragma unroll
    for (int ct = 0; ct < 16; ++ct) {
#pragma unroll
        for (int r = 0; r < 8; ++r) {
            const int t = t0 + r + 8 * g;
            out[((size_t)t * N_ + n) * C_ + ct * 16 + j] = oacc[ct][r];
        }
    }
}

// --------------- Kernel 4: reduce the 16 t-strip partials into att[n][v] ---------------
__global__ void att_reduce_kernel(const float* __restrict__ part, float* __restrict__ att) {
    const int i = blockIdx.x * blockDim.x + threadIdx.x;  // 0..4095
    float s = 0.f;
#pragma unroll
    for (int k = 0; k < 16; ++k) s += part[(size_t)k * (N_ * MV_) + i];
    att[i] = s;
}

extern "C" void kernel_launch(void* const* d_in, const int* in_sizes, int n_in,
                              void* d_out, int out_size, void* d_ws, size_t ws_size,
                              hipStream_t stream) {
    (void)in_sizes; (void)n_in; (void)out_size; (void)ws_size;
    const float*         P    = (const float*)d_in[0];
    const float*         Mm   = (const float*)d_in[1];
    const unsigned char* mask = (const unsigned char*)d_in[2];  // jax bool -> 1 byte
    const float*         Wq   = (const float*)d_in[3];
    const float*         bq   = (const float*)d_in[4];
    const float*         Wk   = (const float*)d_in[5];
    const float*         bk   = (const float*)d_in[6];
    const float*         Wv   = (const float*)d_in[7];
    const float*         bv   = (const float*)d_in[8];

    float* out = (float*)d_out;                       // (T, N, C)
    float* att = out + (size_t)T_ * N_ * C_;          // (N, MV)

    __hip_bfloat16* Qb   = (__hip_bfloat16*)d_ws;                 // 8 MB
    __hip_bfloat16* Kb   = Qb + (size_t)N_ * T_ * C_;             // 64 MB
    __hip_bfloat16* Vtb  = Kb + (size_t)N_ * S_ * C_;             // 64 MB (transposed)
    float*          attp = (float*)(Vtb + (size_t)N_ * C_ * S_);  // 16 strips x 4096

    proj_q_kernel <<<(1024 * 16) / 4, 128, 0, stream>>>(P, Wq, bq, Qb);
    proj_kv_kernel<<<(8192 * 16) / 4, 128, 0, stream>>>(Mm, Wk, bk, Wv, bv, Kb, Vtb);
    attn_kernel   <<<dim3(N_, T_ / 64), 128, 0, stream>>>(Qb, Kb, Vtb, mask, out, attp);
    att_reduce_kernel<<<16, 256, 0, stream>>>(attp, att);
}
